// SpatioTemporalGNN_58841051955385
// MI455X (gfx1250) — matrix operand order repaired and checked by reading, not compiled, not run on hardware
//
#include <hip/hip_runtime.h>
#include <hip/hip_bf16.h>

// ---- problem constants --------------------------------------------------
#define B_   16
#define N_   500
#define T_   256
#define FIN  8
#define H_   64
#define P_   4096                      // B_*T_ combined (b,t) position index
#define EPSV 1e-5f
static constexpr size_t HPc = (size_t)H_ * P_;   // 262144 columns of adjacency GEMM

typedef __attribute__((ext_vector_type(16))) __bf16 v16bf;
typedef __attribute__((ext_vector_type(8)))  __bf16 bf16x8;
typedef __attribute__((ext_vector_type(8)))  float  v8f;
typedef __attribute__((ext_vector_type(4)))  unsigned int u32x4;
typedef __attribute__((ext_vector_type(8)))  int i32x8;
typedef __attribute__((ext_vector_type(4)))  int i32x4;
typedef __attribute__((address_space(1))) i32x4 as1_i32x4;   // global 128b chunk
typedef __attribute__((address_space(3))) i32x4 as3_i32x4;   // LDS 128b chunk

// ---- CDNA5 feature probes (compile-safe fallbacks) ----------------------
#if defined(__gfx1250__) && __has_builtin(__builtin_amdgcn_global_load_async_to_lds_b128)
#define HAVE_ASYNC_LDS 1
#else
#define HAVE_ASYNC_LDS 0
#endif

#if defined(__gfx1250__) && __has_builtin(__builtin_amdgcn_tensor_load_to_lds) && \
    __has_builtin(__builtin_amdgcn_s_wait_tensorcnt)
#define HAVE_TDM 1
#if __has_include(<hip/amd_detail/amd_gfx1250_TDM.h>)
#define TDM_6ARG 1
#else
#define TDM_6ARG 0
#endif
#else
#define HAVE_TDM 0
#endif

static __device__ inline void wait_asynccnt0() {
#if defined(__gfx1250__)
#if __has_builtin(__builtin_amdgcn_s_wait_asynccnt)
  __builtin_amdgcn_s_wait_asynccnt(0);
#else
  asm volatile("s_wait_asynccnt 0" ::: "memory");
#endif
#endif
}

static __device__ inline v16bf cat8(bf16x8 lo, bf16x8 hi) {
  return __builtin_shufflevector(lo, hi, 0,1,2,3,4,5,6,7,8,9,10,11,12,13,14,15);
}
static __device__ inline bf16x8 zero8() {
  bf16x8 z;
  #pragma unroll
  for (int e = 0; e < 8; ++e) z[e] = (__bf16)0.0f;
  return z;
}
static __device__ inline v8f zeroAcc() {
  v8f a;
  #pragma unroll
  for (int e = 0; e < 8; ++e) a[e] = 0.0f;
  return a;
}

// =========================================================================
// Prep: adjacency -> bf16 padded 512x512; W^T -> bf16; conv_w2 -> [h][dk*64+c]
// =========================================================================
__global__ __launch_bounds__(256) void k_prep(
    const float* __restrict__ adj, const float* __restrict__ gw1,
    const float* __restrict__ gw2, const float* __restrict__ cw2,
    __bf16* __restrict__ adjb, __bf16* __restrict__ W1T,
    __bf16* __restrict__ W2T, __bf16* __restrict__ A2)
{
  int tid = blockIdx.x * 256 + threadIdx.x;                  // 262144 threads
  if (tid < 512 * 512) {
    int m = tid >> 9, k = tid & 511;
    adjb[tid] = (m < N_ && k < N_) ? (__bf16)adj[m * N_ + k] : (__bf16)0.0f;
  }
  if (tid < H_ * H_) {
    int h = tid >> 6, c = tid & 63;
    W1T[h * 64 + c] = (__bf16)gw1[c * 64 + h];
    W2T[h * 64 + c] = (__bf16)gw2[c * 64 + h];
  }
  if (tid < H_ * 192) {
    int h = tid / 192, r = tid % 192;
    int dk = r >> 6, c = r & 63;
    A2[h * 192 + dk * 64 + c] = (__bf16)cw2[h * 192 + c * 3 + dk];
  }
}

// =========================================================================
// Layer-1 temporal conv (K contraction = 24, VALU): x(B,N,T,F) -> X1[n][h][p]
// =========================================================================
__global__ __launch_bounds__(256) void k_conv1(
    const float* __restrict__ x, const float* __restrict__ w1,
    const float* __restrict__ b1, __bf16* __restrict__ X1)
{
  size_t idx = (size_t)blockIdx.x * 256 + threadIdx.x;       // 131,072,000
  int p = (int)(idx & 4095);
  int h = (int)((idx >> 12) & 63);
  int n = (int)(idx >> 18);
  int b = p >> 8, t = p & 255;
  float acc = b1[h];
  const float* wr = w1 + h * (FIN * 3);
  size_t xb = ((size_t)b * N_ + n) * T_;
  #pragma unroll
  for (int dk = 0; dk < 3; ++dk) {
    int tt = t + dk - 1;
    if (tt >= 0 && tt < T_) {
      const float* xr = x + (xb + tt) * FIN;
      #pragma unroll
      for (int c = 0; c < FIN; ++c) acc += wr[c * 3 + dk] * xr[c];
    }
  }
  acc = acc > 0.0f ? acc : 0.0f;
  __builtin_nontemporal_store((__bf16)acc, &X1[idx]);
}

// =========================================================================
// Per-node GEMM (M=64):  out[n][m][p] = sum_r A[m][r] * B[r][p]
//   nDk=1 : r = c          (GCN channel matmul, no shift)
//   nDk=3 : r = dk*64 + c  (temporal conv as shifted-column GEMM, halo in LDS)
// A block staged via TDM (tensor_load_to_lds) when available; padded LDS
// mapping off(g) = g + (g>>6)*8 (TDM pad_interval=32dw, pad_amount=4dw).
// =========================================================================
#define LDB_N 72    // padded halves per ldsB row
#define PADH(g) ((g) + (((g) >> 6) << 3))

__global__ __launch_bounds__(256) void k_node_gemm(
    const __bf16* __restrict__ A,    // [64][nDk*64] bf16
    const __bf16* __restrict__ X,    // [500][64][4096] bf16
    __bf16* __restrict__ out,        // [500][64][4096] bf16
    int nDk, const float* __restrict__ bias, int relu)
{
  __shared__ __bf16 ldsA[PADH(64 * 192)];   // 13824 halves = 27.0 KB (max)
  __shared__ __bf16 ldsB[132 * LDB_N];      // 19.0 KB (cols p0-1..p0+128, [j][c])
  const int tid = threadIdx.x, l = tid & 31, w = tid >> 5;
  const int n  = blockIdx.y;
  const int p0 = blockIdx.x * 128;
  const int ldaH = nDk * 64;
  const __bf16* Xn = X + (size_t)n * H_ * P_;

  // ---- stage A: whole [64][ldaH] bf16 block into padded LDS ----
#if HAVE_TDM
  if (w == 0) {
    unsigned ldsOff = (unsigned)(size_t)(__attribute__((address_space(3))) char*)&ldsA[0];
    unsigned long long ga = (unsigned long long)(const void*)A;
    u32x4 g0;
    g0[0] = 1u;                                           // count=1, user mode
    g0[1] = ldsOff;                                       // lds_addr (bytes)
    g0[2] = (unsigned)(ga & 0xFFFFFFFFu);                 // global_addr lo
    g0[3] = (unsigned)((ga >> 32) & 0x1FFFFFFu) | (2u << 30);  // hi + type=2
    i32x8 g1;
    g1[0] = (1 << 16) | (1 << 20) | (4 << 22) | (3 << 25); // 2B elems, pad 32dw+4dw
    g1[1] = (ldaH & 0xFFFF) << 16;                        // tensor_dim0 lo16
    g1[2] = (64 & 0xFFFF) << 16;                          // dim0 hi=0 | tensor_dim1
    g1[3] = (ldaH << 16);                                 // dim1 hi=0 | tile_dim0
    g1[4] = 64;                                           // tile_dim1 | tile_dim2=0
    g1[5] = ldaH;                                         // tensor_dim0_stride lo32
    g1[6] = 0; g1[7] = 0;
    i32x4 gz; gz[0] = gz[1] = gz[2] = gz[3] = 0;
#if TDM_6ARG
    i32x8 gz8;
    #pragma unroll
    for (int e = 0; e < 8; ++e) gz8[e] = 0;
    __builtin_amdgcn_tensor_load_to_lds(g0, g1, gz, gz, gz8, 0);
#else
    __builtin_amdgcn_tensor_load_to_lds(g0, g1, gz, gz, 0);
#endif
    __builtin_amdgcn_s_wait_tensorcnt(0);
  }
#else
  {
    const int rowChunks = ldaH >> 3;               // 16B chunks per row (8 or 24)
    for (int c = tid; c < 64 * rowChunks; c += 256) {
      int hr = c / rowChunks, cb = c % rowChunks;
      int g = hr * ldaH + cb * 8;
      *(bf16x8*)&ldsA[PADH(g)] = *(const bf16x8*)(A + g);
    }
  }
#endif
  // ---- stage B transposed: ldsB[j][c] = X[c][p0-1+j], j = 1..128 main ----
  #pragma unroll
  for (int r = 0; r < 4; ++r) {
    int chunk = tid + 256 * r;                     // 1024 chunks = 64c x 16
    int c = chunk >> 4, cc = chunk & 15;
    bf16x8 v = *(const bf16x8*)(Xn + (size_t)c * P_ + p0 + cc * 8);
    #pragma unroll
    for (int e = 0; e < 8; ++e) ldsB[(cc * 8 + e + 1) * LDB_N + c] = v[e];
  }
  // ---- halos j=0 (p0-1) and j=129 (p0+128), zero across batch segments ----
  if (tid < 128) {
    int c = tid & 63, side = tid >> 6;
    int j = side ? 129 : 0;
    int p = side ? (p0 + 128) : (p0 - 1);
    bool valid = side ? ((p0 & 255) == 0) : ((p0 & 255) == 128);
    __bf16 v = (__bf16)0.0f;
    if (valid) v = Xn[(size_t)c * P_ + p];
    ldsB[j * LDB_N + c] = v;
  }
  __syncthreads();

  v8f acc[4];
  #pragma unroll
  for (int mt = 0; mt < 4; ++mt) acc[mt] = zeroAcc();

  const int kbA  = (l >= 16) ? 8 : 0;    // interleaved A K-striping
  const int kbB  = (l >= 16) ? 8 : 0;    // interleaved B K-striping
  const int colL = w * 16 + (l & 15);    // local output column
  const int jbase = (nDk == 1) ? 1 : 0;  // no-shift GEMM reads j = col+1

  for (int dk = 0; dk < nDk; ++dk) {
    const __bf16* brow = &ldsB[(colL + dk + jbase) * LDB_N];
    #pragma unroll
    for (int kb = 0; kb < 2; ++kb) {
      const __bf16* pb = brow + kb * 32 + kbB;
      v16bf bfr = cat8(*(const bf16x8*)pb, *(const bf16x8*)(pb + 16));
      int r0 = dk * 64 + kb * 32;
      #pragma unroll
      for (int mt = 0; mt < 4; ++mt) {
        const __bf16* pa = &ldsA[PADH((mt * 16 + (l & 15)) * ldaH + r0 + kbA)];
        v16bf afr = cat8(*(const bf16x8*)pa, *(const bf16x8*)(pa + 16));
        acc[mt] = __builtin_amdgcn_wmma_f32_16x16x32_bf16(
            false, afr, false, bfr, (short)0, acc[mt], false, false);
      }
    }
  }

  // ---- epilogue ----
  #pragma unroll
  for (int mt = 0; mt < 4; ++mt) {
    #pragma unroll
    for (int i = 0; i < 8; ++i) {
      int m = mt * 16 + i + ((l >= 16) ? 8 : 0);
      float v = acc[mt][i];
      if (bias) v += bias[m];
      if (relu) v = v > 0.0f ? v : 0.0f;
      __builtin_nontemporal_store((__bf16)v,
          &out[((size_t)n * H_ + m) * P_ + p0 + colL]);
    }
  }
}

// =========================================================================
// Adjacency GEMM: C(500x262144) = adjb(512x512) @ S(500x262144)
// 128x128 block tile, K-step 32, LDS double buffered.
// A tile copied global->LDS via GLOBAL_LOAD_ASYNC_TO_LDS_B128 (ASYNCcnt)
// when available. Fused epilogue: + gcn_bias -> relu -> batchnorm -> bf16
// =========================================================================
#define ADJ_LDA 40   // padded halves per LDS row (32 data + 8 pad)

__global__ __launch_bounds__(256) void k_adj_gemm(
    const __bf16* __restrict__ A,    // [512][512] bf16 (zero padded)
    const __bf16* __restrict__ Bm,   // [500][262144] bf16
    __bf16* __restrict__ out,        // [500][262144] bf16
    const float* __restrict__ gb,  const float* __restrict__ bg,
    const float* __restrict__ bbe, const float* __restrict__ bm,
    const float* __restrict__ bv)
{
  __shared__ __bf16 ldsA[2][128 * ADJ_LDA];   // 2 x 10 KB, rows m, [m][k]
  __shared__ __bf16 ldsB[2][128 * ADJ_LDA];   // 2 x 10 KB, transposed [q][k]
  const int tid = threadIdx.x, l = tid & 31, w = tid >> 5;
  const int wm = w & 3;           // 32-row slice
  const int wn = w >> 2;          // 64-col slice
  const int m0 = blockIdx.x * 128;
  const size_t q0 = (size_t)blockIdx.y * 128;

  v8f acc[2][4];
  #pragma unroll
  for (int mt = 0; mt < 2; ++mt)
    #pragma unroll
    for (int nt = 0; nt < 4; ++nt) acc[mt][nt] = zeroAcc();

  bf16x8 rb[2];
#if !HAVE_ASYNC_LDS
  bf16x8 ra[2];
  auto loadA_regs = [&](int kk) {
    const int k0 = kk * 32;
    #pragma unroll
    for (int r = 0; r < 2; ++r) {              // A: 128 rows x 4 chunks
      int chunk = tid + 256 * r;
      int row = chunk >> 2, cb = chunk & 3;
      ra[r] = *(const bf16x8*)(A + (size_t)(m0 + row) * 512 + k0 + cb * 8);
    }
  };
  auto storeA = [&](int buf) {
    #pragma unroll
    for (int r = 0; r < 2; ++r) {
      int chunk = tid + 256 * r;
      int row = chunk >> 2, cb = chunk & 3;
      *(bf16x8*)&ldsA[buf][row * ADJ_LDA + cb * 8] = ra[r];
    }
  };
#else
  auto issueA = [&](int kk, int buf) {         // async copy straight into LDS
    const int k0 = kk * 32;
    #pragma unroll
    for (int r = 0; r < 2; ++r) {
      int chunk = tid + 256 * r;
      int row = chunk >> 2, cb = chunk & 3;
      __builtin_amdgcn_global_load_async_to_lds_b128(
          (as1_i32x4*)(const void*)
              (A + (size_t)(m0 + row) * 512 + k0 + cb * 8),
          (as3_i32x4*)(void*)
              &ldsA[buf][row * ADJ_LDA + cb * 8],
          0, 0);
    }
  };
#endif
  auto loadB_regs = [&](int kk) {
    const int k0 = kk * 32;
    #pragma unroll
    for (int r = 0; r < 2; ++r) {              // B: 32 k-rows x 16 chunks
      int chunk = tid + 256 * r;
      int kr = chunk >> 4, cc = chunk & 15;
      int krow = k0 + kr;
      rb[r] = (krow < N_) ? *(const bf16x8*)(Bm + (size_t)krow * HPc + q0 + cc * 8)
                          : zero8();
    }
  };
  auto storeB = [&](int buf) {                 // transpose scatter [q][k]
    #pragma unroll
    for (int r = 0; r < 2; ++r) {
      int chunk = tid + 256 * r;
      int kr = chunk >> 4, cc = chunk & 15;
      #pragma unroll
      for (int e = 0; e < 8; ++e)
        ldsB[buf][(cc * 8 + e) * ADJ_LDA + kr] = rb[r][e];
    }
  };
  auto compute = [&](int buf) {
    const int kbA = (l >= 16) ? 8 : 0;
    const int kbB = (l >= 16) ? 8 : 0;
    v16bf bfr[4];
    #pragma unroll
    for (int nt = 0; nt < 4; ++nt) {
      int col = wn * 64 + nt * 16 + (l & 15);
      const __bf16* p = &ldsB[buf][col * ADJ_LDA + kbB];
      bfr[nt] = cat8(*(const bf16x8*)p, *(const bf16x8*)(p + 16));
    }
    #pragma unroll
    for (int mt = 0; mt < 2; ++mt) {
      int row = wm * 32 + mt * 16 + (l & 15);
      const __bf16* p = &ldsA[buf][row * ADJ_LDA + kbA];
      v16bf afr = cat8(*(const bf16x8*)p, *(const bf16x8*)(p + 16));
      #pragma unroll
      for (int nt = 0; nt < 4; ++nt)
        acc[mt][nt] = __builtin_amdgcn_wmma_f32_16x16x32_bf16(
            false, afr, false, bfr[nt], (short)0, acc[mt][nt], false, false);
    }
  };

  loadB_regs(0);
#if HAVE_ASYNC_LDS
  issueA(0, 0);
#else
  loadA_regs(0);
#endif
  for (int kk = 0; kk < 16; ++kk) {
    int buf = kk & 1;
    storeB(buf);
#if HAVE_ASYNC_LDS
    wait_asynccnt0();                // A tile for this iter landed in LDS
#else
    storeA(buf);
#endif
    __syncthreads();                 // single barrier: safe w/ double buffer
    if (kk + 1 < 16) {
      loadB_regs(kk + 1);
#if HAVE_ASYNC_LDS
      issueA(kk + 1, buf ^ 1);
#else
      loadA_regs(kk + 1);
#endif
    }
    if (kk + 2 < 16) {               // L2 prefetch of next-next K tile
      __builtin_prefetch(A + (size_t)(m0 + (tid & 127)) * 512 + (kk + 2) * 32, 0, 1);
      int krow = (kk + 2) * 32 + (tid & 31);
      if (krow < N_)
        __builtin_prefetch(Bm + (size_t)krow * HPc + q0 + ((tid >> 5) & 7) * 16, 0, 1);
    }
    compute(buf);
  }

  // ---- fused epilogue: +bias, relu, batchnorm(eval), store bf16 ----
  #pragma unroll
  for (int mt = 0; mt < 2; ++mt) {
    #pragma unroll
    for (int i = 0; i < 8; ++i) {
      int m = m0 + wm * 32 + mt * 16 + i + ((l >= 16) ? 8 : 0);
      if (m < N_) {
        #pragma unroll
        for (int nt = 0; nt < 4; ++nt) {
          size_t q = q0 + wn * 64 + nt * 16 + (l & 15);
          int h = (int)(q >> 12);
          float v = acc[mt][nt][i] + gb[h];
          v = v > 0.0f ? v : 0.0f;
          float sc = bg[h] / sqrtf(bv[h] + EPSV);
          v = (v - bm[h]) * sc + bbe[h];
          __builtin_nontemporal_store((__bf16)v, &out[(size_t)m * HPc + q]);
        }
      }
    }
  }
}

// =========================================================================
// Head: mean over T, then MLP 64 -> 32 (relu) -> 1.  One block per (b,n).
// =========================================================================
__global__ __launch_bounds__(256) void k_head(
    const __bf16* __restrict__ Y, const float* __restrict__ w1,
    const float* __restrict__ b1, const float* __restrict__ w2,
    const float* __restrict__ b2, float* __restrict__ out)
{
  __shared__ float part[256];
  __shared__ float accs[64];
  __shared__ float zs[32];
  const int bid = blockIdx.x;          // 8000 = b*500+n
  const int b = bid / N_, n = bid % N_;
  const int tid = threadIdx.x;
  const int h = tid >> 2, qt = tid & 3;
  const __bf16* yr = Y + ((size_t)n * H_ + h) * P_ + b * T_ + qt * 64;
  float s = 0.0f;
  for (int i = 0; i < 64; ++i) s += (float)yr[i];
  part[tid] = s;
  __syncthreads();
  if (qt == 0)
    accs[h] = (part[tid] + part[tid + 1] + part[tid + 2] + part[tid + 3]) * (1.0f / 256.0f);
  __syncthreads();
  if (tid < 32) {
    float z = b1[tid];
    for (int c = 0; c < 64; ++c) z += accs[c] * w1[c * 32 + tid];
    zs[tid] = z > 0.0f ? z : 0.0f;
  }
  __syncthreads();
  if (tid == 0) {
    float o = b2[0];
    for (int j = 0; j < 32; ++j) o += zs[j] * w2[j];
    out[bid] = o;
  }
}

// =========================================================================
extern "C" void kernel_launch(void* const* d_in, const int* in_sizes, int n_in,
                              void* d_out, int out_size, void* d_ws, size_t ws_size,
                              hipStream_t stream)
{
  const float* x    = (const float*)d_in[0];
  const float* adj  = (const float*)d_in[1];
  const float* cw1  = (const float*)d_in[2];
  const float* cb1  = (const float*)d_in[3];
  const float* cw2  = (const float*)d_in[4];
  const float* cb2  = (const float*)d_in[5];
  const float* gw1  = (const float*)d_in[6];
  const float* gb1  = (const float*)d_in[7];
  const float* gw2  = (const float*)d_in[8];
  const float* gb2  = (const float*)d_in[9];
  const float* bg1  = (const float*)d_in[10];
  const float* bbe1 = (const float*)d_in[11];
  const float* bm1  = (const float*)d_in[12];
  const float* bv1  = (const float*)d_in[13];
  const float* bg2  = (const float*)d_in[14];
  const float* bbe2 = (const float*)d_in[15];
  const float* bm2  = (const float*)d_in[16];
  const float* bv2  = (const float*)d_in[17];
  const float* ow1  = (const float*)d_in[18];
  const float* ob1  = (const float*)d_in[19];
  const float* ow2  = (const float*)d_in[20];
  const float* ob2  = (const float*)d_in[21];
  float* out = (float*)d_out;

  char* ws = (char*)d_ws;
  const size_t NB = (size_t)N_ * H_ * P_ * sizeof(__bf16);   // 262,144,000 B
  __bf16* buf0 = (__bf16*)(ws);             // activations (rotating)
  __bf16* buf1 = (__bf16*)(ws + NB);
  __bf16* buf2 = (__bf16*)(ws + 2 * NB);
  __bf16* adjb = (__bf16*)(ws + 3 * NB);    // 512x512 bf16
  __bf16* W1T  = adjb + 512 * 512;
  __bf16* W2T  = W1T + 64 * 64;
  __bf16* A2   = W2T + 64 * 64;             // [64][192]

  k_prep<<<1024, 256, 0, stream>>>(adj, gw1, gw2, cw2, adjb, W1T, W2T, A2);
  // layer 1
  k_conv1<<<512000, 256, 0, stream>>>(x, cw1, cb1, buf0);
  k_node_gemm<<<dim3(32, N_), 256, 0, stream>>>(W1T, buf0, buf1, 1, nullptr, 0);
  k_adj_gemm<<<dim3(4, 2048), 256, 0, stream>>>(adjb, buf1, buf2, gb1, bg1, bbe1, bm1, bv1);
  // layer 2 (conv as shifted GEMM, fused bias+relu)
  k_node_gemm<<<dim3(32, N_), 256, 0, stream>>>(A2, buf2, buf0, 3, cb2, 1);
  k_node_gemm<<<dim3(32, N_), 256, 0, stream>>>(W2T, buf0, buf1, 1, nullptr, 0);
  k_adj_gemm<<<dim3(4, 2048), 256, 0, stream>>>(adjb, buf1, buf2, gb2, bg2, bbe2, bm2, bv2);
  // head
  k_head<<<8000, 256, 0, stream>>>(buf2, ow1, ob1, ow2, ob2, out);
}
